// TSA_68719476736041
// MI455X (gfx1250) — matrix operand order repaired and checked by reading, not compiled, not run on hardware
//
#include <hip/hip_runtime.h>

typedef __attribute__((ext_vector_type(16))) __bf16 v16bf;
typedef __attribute__((ext_vector_type(8)))  __bf16 v8bf;
typedef __attribute__((ext_vector_type(8)))  float  v8f;
typedef __attribute__((ext_vector_type(4)))  float  v4f;
typedef __bf16 bf16;

#define B_  16
#define D_  512
#define C_  256
#define T_  2048
#define PB  ((size_t)B_ * C_ * T_)   // 8,388,608 elements per bf16 buffer

__device__ __forceinline__ v8f wmma_bf(const v16bf& a, const v16bf& b, v8f c) {
  return __builtin_amdgcn_wmma_f32_16x16x32_bf16(false, a, false, b, (short)0, c,
                                                 false, false);
}

// A-fragment 16x32 bf16: per lane the 16 K values are one contiguous run
// starting at hi*16 -> two 16B loads when K is contiguous in memory.
__device__ __forceinline__ v16bf load_a_bf16(const bf16* rowbase, int hi) {
  v8bf r0 = *(const v8bf*)(rowbase + hi * 16);
  v8bf r1 = *(const v8bf*)(rowbase + hi * 16 + 8);
  v16bf a;
#pragma unroll
  for (int i = 0; i < 8; ++i) {
    a[i] = r0[i];
    a[8 + i] = r1[i];
  }
  return a;
}

__device__ __forceinline__ v16bf load_a_f32(const float* rowbase, int hi) {
  v4f r0 = *(const v4f*)(rowbase + hi * 16);
  v4f r1 = *(const v4f*)(rowbase + hi * 16 + 4);
  v4f r2 = *(const v4f*)(rowbase + hi * 16 + 8);
  v4f r3 = *(const v4f*)(rowbase + hi * 16 + 12);
  v16bf a;
#pragma unroll
  for (int i = 0; i < 4; ++i) {
    a[i] = (bf16)r0[i];
    a[4 + i] = (bf16)r1[i];
    a[8 + i] = (bf16)r2[i];
    a[12 + i] = (bf16)r3[i];
  }
  return a;
}

// B-fragment 32x16 bf16 from an LDS tile stored TRANSPOSED as [t][d] with
// padded row stride RS (elements): per lane (col n) the 16 K values are one
// contiguous run starting at hi*16 -> two ds_load_b128.
template <int RS>
__device__ __forceinline__ v16bf load_b_lds(const bf16* tile, int trow, int hi) {
  const bf16* p = tile + trow * RS + hi * 16;
  v8bf r0 = *(const v8bf*)(p);
  v8bf r1 = *(const v8bf*)(p + 8);
  v16bf b;
#pragma unroll
  for (int i = 0; i < 8; ++i) {
    b[i] = r0[i];
    b[8 + i] = r1[i];
  }
  return b;
}

// ---------------------------------------------------------------------------
// K1: fused projection GEMM. Y[768, T] = [W1;W2;W3][768,512] @ x_b[512, T]
// o<256 -> x0t TRANSPOSED [b][t][c] pre-act (bias cancels in BN),
// 256..511 -> x1 pre-act [b][c][t], 512..767 -> x2 (+b3) [b][c][t].
// Block = 4 waves covering 64 o x 64 t; the x B-tile (32d x 64t) is staged
// once per k-step into LDS (transposed, bf16) and shared by all 4 waves.
// ---------------------------------------------------------------------------
__global__ __launch_bounds__(128) void k_proj(
    const float* __restrict__ x, const float* __restrict__ W1,
    const float* __restrict__ W2, const float* __restrict__ W3,
    const float* __restrict__ b3, bf16* __restrict__ x0t,
    bf16* __restrict__ x1r, bf16* __restrict__ x2b) {
  const int lane = threadIdx.x & 31;
  const int wave = threadIdx.x >> 5;
  const int row = lane & 15, hi = lane >> 4, n = lane & 15;
  const int b = blockIdx.z;
  const int o0 = blockIdx.y * 64 + wave * 16;
  const int t0 = blockIdx.x * 64;

  const int o = o0 + row;
  const float* Wrow;
  if (o < 256)      Wrow = W1 + (size_t)o * 512;
  else if (o < 512) Wrow = W2 + (size_t)(o - 256) * 512;
  else              Wrow = W3 + (size_t)(o - 512) * 512;

  const float* xb = x + (size_t)b * D_ * T_;

  constexpr int RS = 40;  // padded row stride: 40*2B = 80B (16B multiple)
  __shared__ __align__(16) bf16 Bt[64 * RS];  // [t][d] transposed tile

  v8f acc[4] = {{}, {}, {}, {}};
  for (int kk = 0; kk < 512; kk += 32) {
    __syncthreads();  // previous iteration's reads done
    // stage x[kk..kk+31][t0..t0+63] -> Bt[t][d] (coalesced fp32 reads)
#pragma unroll
    for (int e = threadIdx.x; e < 32 * 64; e += 128) {
      int d = e >> 6, t = e & 63;
      Bt[t * RS + d] = (bf16)xb[(size_t)(kk + d) * T_ + t0 + t];
    }
    __syncthreads();
    v16bf af = load_a_f32(Wrow + kk, hi);
#pragma unroll
    for (int j = 0; j < 4; ++j)
      acc[j] = wmma_bf(af, load_b_lds<RS>(Bt, j * 16 + n, hi), acc[j]);
  }
#pragma unroll
  for (int j = 0; j < 4; ++j) {
#pragma unroll
    for (int v = 0; v < 8; ++v) {
      int oo = o0 + v + hi * 8;
      int t = t0 + j * 16 + n;
      float val = acc[j][v];
      if (oo < 256)
        x0t[((size_t)b * T_ + t) * C_ + oo] = (bf16)val;  // transposed
      else if (oo < 512)
        x1r[((size_t)b * C_ + (oo - 256)) * T_ + t] = (bf16)val;
      else
        x2b[((size_t)b * C_ + (oo - 512)) * T_ + t] = (bf16)(val + b3[oo - 512]);
    }
  }
}

// ---------------------------------------------------------------------------
// K2: per-channel BN stats over (B,T) -> fused scale/shift. 512 blocks.
// ---------------------------------------------------------------------------
__global__ __launch_bounds__(256) void k_bnstats(
    const bf16* __restrict__ x0t, const bf16* __restrict__ x1r,
    const float* __restrict__ g1, const float* __restrict__ be1,
    const float* __restrict__ g2, const float* __restrict__ be2,
    float* __restrict__ bnscale, float* __restrict__ bnshift) {
  const int c = blockIdx.x;  // 0..511
  const int cl = c & 255;
  float s = 0.f, sq = 0.f;
  for (int i = threadIdx.x; i < B_ * T_; i += 256) {
    float v;
    if (c < 256)
      v = (float)x0t[(size_t)i * C_ + cl];
    else
      v = (float)x1r[((size_t)(i >> 11) * C_ + cl) * T_ + (i & (T_ - 1))];
    s += v;
    sq += v * v;
  }
  __shared__ float ls[256], lq[256];
  ls[threadIdx.x] = s;
  lq[threadIdx.x] = sq;
  __syncthreads();
  for (int off = 128; off > 0; off >>= 1) {
    if (threadIdx.x < (unsigned)off) {
      ls[threadIdx.x] += ls[threadIdx.x + off];
      lq[threadIdx.x] += lq[threadIdx.x + off];
    }
    __syncthreads();
  }
  if (threadIdx.x == 0) {
    const float inv = 1.0f / (float)(B_ * T_);
    float mean = ls[0] * inv;
    float var = lq[0] * inv - mean * mean;
    float g = (c < 256) ? g1[cl] : g2[cl];
    float be = (c < 256) ? be1[cl] : be2[cl];
    float sc = g * rsqrtf(var + 1e-5f);
    bnscale[c] = sc;
    bnshift[c] = be - mean * sc;
  }
}

// ---------------------------------------------------------------------------
// K3: apply BN + ReLU in place on x0t / x1r.
// ---------------------------------------------------------------------------
__global__ __launch_bounds__(256) void k_bnrelu(
    bf16* __restrict__ x0t, bf16* __restrict__ x1r,
    const float* __restrict__ bnscale, const float* __restrict__ bnshift) {
  const size_t NTOT = 2 * PB;
  for (size_t idx = (size_t)blockIdx.x * 256 + threadIdx.x; idx < NTOT;
       idx += (size_t)gridDim.x * 256) {
    int buf = idx >= PB;
    size_t j = idx - (buf ? PB : 0);
    int gc;
    bf16* p;
    if (buf) {
      gc = 256 + (int)((j >> 11) & 255);  // x1r: [b][c][t]
      p = x1r;
    } else {
      gc = (int)(j & 255);                // x0t: [b][t][c]
      p = x0t;
    }
    float v = (float)p[j];
    v = v * bnscale[gc] + bnshift[gc];
    p[j] = (bf16)(v > 0.f ? v : 0.f);
  }
}

// ---------------------------------------------------------------------------
// K4a: per-row softmax stats. L[k,t] = sum_c x0t[k,c] x1[c,t]. Block =
// (b, 16 k-rows); A-fragments hoisted (k fixed); 8 waves stream t tiles,
// staging each x1 tile to LDS with async global->LDS loads (ASYNCcnt).
// ---------------------------------------------------------------------------
__global__ __launch_bounds__(256) void k_rowstats(
    const bf16* __restrict__ x0t, const bf16* __restrict__ x1,
    float* __restrict__ mrow, float* __restrict__ lrow) {
  const int lane = threadIdx.x & 31;
  const int wave = threadIdx.x >> 5;
  const int row = lane & 15, hi = lane >> 4, n = lane & 15;
  const int b = blockIdx.y;
  const int k0 = blockIdx.x * 16;
  const bf16* x1b = x1 + (size_t)b * C_ * T_;

  // hoisted A fragments: rows k0+row fixed for the whole block
  v16bf af[8];
  const bf16* arow = x0t + ((size_t)b * T_ + k0 + row) * C_;
#pragma unroll
  for (int ci = 0; ci < 8; ++ci) af[ci] = load_a_bf16(arow + ci * 32, hi);

  __shared__ __align__(16) bf16 S1[8][C_ * 16];  // 8 KB per wave
  const unsigned ldsbase = (unsigned)(size_t)(&S1[wave][0]);
  const int cq = lane >> 1, half = lane & 1;

  float rm[8], rl[8];
#pragma unroll
  for (int v = 0; v < 8; ++v) {
    rm[v] = -3.0e38f;
    rl[v] = 0.f;
  }

  for (int tt = wave; tt < T_ / 16; tt += 8) {
    const int t0 = tt * 16;
    // async-stage x1[0..255][t0..t0+15] (row-major [c][16t]) into S1[wave]
#pragma unroll
    for (int it = 0; it < 16; ++it) {
      int c = it * 16 + cq;
      unsigned laddr = ldsbase + (unsigned)(c * 32 + half * 16);
      unsigned long long gaddr =
          (unsigned long long)(size_t)(x1b + (size_t)c * T_ + t0 + half * 8);
      asm volatile("global_load_async_to_lds_b128 %0, %1, off" ::"v"(laddr),
                   "v"(gaddr)
                   : "memory");
    }
    asm volatile("s_wait_asynccnt 0x0" ::: "memory");

    v8f s = {};
    for (int c = 0; c < C_; c += 32) {
      v16bf bfv;
#pragma unroll
      for (int i = 0; i < 16; ++i)
        bfv[i] = S1[wave][(c + i + hi * 16) * 16 + n];
      s = wmma_bf(af[c >> 5], bfv, s);
    }
#pragma unroll
    for (int v = 0; v < 8; ++v) {
      float val = s[v];
      if (val > rm[v]) {
        rl[v] = rl[v] * __expf(rm[v] - val) + 1.0f;
        rm[v] = val;
      } else {
        rl[v] += __expf(val - rm[v]);
      }
    }
  }
  // reduce across the 16 lanes that share the same rows
#pragma unroll
  for (int off = 1; off < 16; off <<= 1) {
#pragma unroll
    for (int v = 0; v < 8; ++v) {
      float om = __shfl_xor(rm[v], off, 32);
      float ol = __shfl_xor(rl[v], off, 32);
      float M = fmaxf(rm[v], om);
      rl[v] = rl[v] * __expf(rm[v] - M) + ol * __expf(om - M);
      rm[v] = M;
    }
  }
  __shared__ float pm[8][16], pl[8][16];
  if ((lane & 15) == 0) {
#pragma unroll
    for (int v = 0; v < 8; ++v) {
      pm[wave][v + hi * 8] = rm[v];
      pl[wave][v + hi * 8] = rl[v];
    }
  }
  __syncthreads();
  if (threadIdx.x < 16) {
    int r = threadIdx.x;
    float M = pm[0][r], L = pl[0][r];
#pragma unroll
    for (int w = 1; w < 8; ++w) {
      float om = pm[w][r], ol = pl[w][r];
      float nm = fmaxf(M, om);
      L = L * __expf(M - nm) + ol * __expf(om - nm);
      M = nm;
    }
    mrow[(size_t)b * T_ + k0 + r] = M;
    lrow[(size_t)b * T_ + k0 + r] = L;
  }
}

// ---------------------------------------------------------------------------
// K4b: M[c, t_tile] = sum_k x2[c,k] * P[k,t]. Block = (b, 16-col t tile),
// 8 waves. x1 B-fragments hoisted (t fixed). Per 128-row k block: each wave
// recomputes one 16x16 logit tile, exponentiates into LDS TRANSPOSED as bf16
// Pt[t][k] (so PV B-frags are two ds_load_b128), then all waves accumulate
// their 32 channels with x2 @ P (A-frags = two b128 global loads).
// ---------------------------------------------------------------------------
__global__ __launch_bounds__(256) void k_attn(
    const bf16* __restrict__ x0t, const bf16* __restrict__ x1,
    const bf16* __restrict__ x2, const float* __restrict__ mrow,
    const float* __restrict__ lrow, bf16* __restrict__ Mbuf) {
  const int lane = threadIdx.x & 31;
  const int wave = threadIdx.x >> 5;
  const int row = lane & 15, hi = lane >> 4, n = lane & 15;
  const int b = blockIdx.y;
  const int t0 = blockIdx.x * 16;
  const bf16* x1b = x1 + (size_t)b * C_ * T_;
  const bf16* x2b = x2 + (size_t)b * C_ * T_;
  const float* mb = mrow + (size_t)b * T_;
  const float* lb = lrow + (size_t)b * T_;

  // hoisted x1 B fragments (t tile fixed for whole block)
  v16bf bx1[8];
#pragma unroll
  for (int ci = 0; ci < 8; ++ci) {
#pragma unroll
    for (int i = 0; i < 16; ++i)
      bx1[ci][i] = x1b[(size_t)(ci * 32 + i + hi * 16) * T_ + t0 + n];
  }

  constexpr int KP = 136;  // padded k-stride (keeps rows 16B aligned)
  __shared__ __align__(16) bf16 Pt[16 * KP];

  const int c0 = wave * 32;
  v8f accA = {}, accB = {};
  const bf16* a2base0 = x2b + (size_t)(c0 + row) * T_;
  const bf16* a2base1 = x2b + (size_t)(c0 + 16 + row) * T_;

  for (int kb = 0; kb < T_; kb += 128) {
    if (kb + 128 < T_) {  // hint next x2 stripe toward the caches
      __builtin_prefetch(a2base0 + kb + 128, 0, 1);
      __builtin_prefetch(a2base1 + kb + 128, 0, 1);
    }
    const int kr = kb + wave * 16;
    v8f s = {};
    const bf16* arow = x0t + ((size_t)b * T_ + kr + row) * C_;
    for (int c = 0; c < C_; c += 32)
      s = wmma_bf(load_a_bf16(arow + c, hi), bx1[c >> 5], s);
#pragma unroll
    for (int v = 0; v < 8; ++v) {
      int k = kr + v + hi * 8;
      float p = __expf(s[v] - mb[k]) / lb[k];
      Pt[n * KP + (wave * 16 + v + hi * 8)] = (bf16)p;  // transposed store
    }
    __syncthreads();
#pragma unroll
    for (int kk = 0; kk < 128; kk += 32) {
      v16bf a1 = load_a_bf16(a2base0 + kb + kk, hi);
      v16bf a2 = load_a_bf16(a2base1 + kb + kk, hi);
      v8bf p0 = *(const v8bf*)(&Pt[n * KP + kk + hi * 16]);
      v8bf p1 = *(const v8bf*)(&Pt[n * KP + kk + hi * 16 + 8]);
      v16bf bp;
#pragma unroll
      for (int i = 0; i < 8; ++i) {
        bp[i] = p0[i];
        bp[8 + i] = p1[i];
      }
      accA = wmma_bf(a1, bp, accA);
      accB = wmma_bf(a2, bp, accB);
    }
    __syncthreads();
  }
#pragma unroll
  for (int v = 0; v < 8; ++v) {
    Mbuf[((size_t)b * C_ + c0 + v + hi * 8) * T_ + t0 + n] = (bf16)accA[v];
    Mbuf[((size_t)b * C_ + c0 + 16 + v + hi * 8) * T_ + t0 + n] = (bf16)accB[v];
  }
}

// ---------------------------------------------------------------------------
// K5: out = W4[512,256] @ M + b4 + x (residual). Block = 4 waves, 64o x 64t;
// the M B-tile (32c x 64t) is staged per k-step into LDS (transposed) and
// shared by all 4 waves; B-frags are two ds_load_b128.
// ---------------------------------------------------------------------------
__global__ __launch_bounds__(128) void k_out(
    const bf16* __restrict__ Mbuf, const float* __restrict__ W4,
    const float* __restrict__ b4, const float* __restrict__ x,
    float* __restrict__ out) {
  const int lane = threadIdx.x & 31;
  const int wave = threadIdx.x >> 5;
  const int row = lane & 15, hi = lane >> 4, n = lane & 15;
  const int b = blockIdx.z;
  const int o0 = blockIdx.y * 64 + wave * 16;
  const int t0 = blockIdx.x * 64;
  const bf16* Mb = Mbuf + (size_t)b * C_ * T_;
  const float* Wrow = W4 + (size_t)(o0 + row) * C_;

  constexpr int RS = 40;
  __shared__ __align__(16) bf16 Mt[64 * RS];  // [t][c] transposed tile

  v8f acc[4] = {{}, {}, {}, {}};
  for (int kk = 0; kk < C_; kk += 32) {
    __syncthreads();
#pragma unroll
    for (int e = threadIdx.x; e < 32 * 64; e += 128) {
      int d = e >> 6, t = e & 63;
      Mt[t * RS + d] = Mb[(size_t)(kk + d) * T_ + t0 + t];
    }
    __syncthreads();
    v16bf af = load_a_f32(Wrow + kk, hi);
#pragma unroll
    for (int j = 0; j < 4; ++j)
      acc[j] = wmma_bf(af, load_b_lds<RS>(Mt, j * 16 + n, hi), acc[j]);
  }
#pragma unroll
  for (int j = 0; j < 4; ++j) {
#pragma unroll
    for (int v = 0; v < 8; ++v) {
      int oo = o0 + v + hi * 8;
      int t = t0 + j * 16 + n;
      size_t idx = ((size_t)b * D_ + oo) * T_ + t;
      out[idx] = acc[j][v] + b4[oo] + x[idx];
    }
  }
}

// ---------------------------------------------------------------------------
extern "C" void kernel_launch(void* const* d_in, const int* in_sizes, int n_in,
                              void* d_out, int out_size, void* d_ws,
                              size_t ws_size, hipStream_t stream) {
  (void)in_sizes; (void)n_in; (void)out_size; (void)ws_size;
  const float* x   = (const float*)d_in[0];
  const float* W1  = (const float*)d_in[1];
  const float* b1  = (const float*)d_in[2]; (void)b1;   // cancels in BN
  const float* W2  = (const float*)d_in[3];
  const float* b2  = (const float*)d_in[4]; (void)b2;   // cancels in BN
  const float* W3  = (const float*)d_in[5];
  const float* b3  = (const float*)d_in[6];
  const float* W4  = (const float*)d_in[7];
  const float* b4  = (const float*)d_in[8];
  const float* g1  = (const float*)d_in[9];
  const float* be1 = (const float*)d_in[10];
  const float* g2  = (const float*)d_in[11];
  const float* be2 = (const float*)d_in[12];
  float* out = (float*)d_out;

  // workspace layout (~67.2 MB)
  bf16* x0t  = (bf16*)d_ws;     // [16,2048,256] bf16 (transposed, A-operand)
  bf16* x1r  = x0t + PB;        // [16,256,2048]
  bf16* x2b  = x1r + PB;        // [16,256,2048]
  bf16* Mbuf = x2b + PB;        // [16,256,2048]
  float* bnscale = (float*)(Mbuf + PB);      // [512]
  float* bnshift = bnscale + 512;            // [512]
  float* mrow    = bnshift + 512;            // [16,2048]
  float* lrow    = mrow + (size_t)B_ * T_;   // [16,2048]

  k_proj<<<dim3(T_ / 64, 768 / 64, B_), 128, 0, stream>>>(x, W1, W2, W3, b3,
                                                          x0t, x1r, x2b);
  k_bnstats<<<512, 256, 0, stream>>>(x0t, x1r, g1, be1, g2, be2, bnscale,
                                     bnshift);
  k_bnrelu<<<8192, 256, 0, stream>>>(x0t, x1r, bnscale, bnshift);
  k_rowstats<<<dim3(T_ / 16, B_), 256, 0, stream>>>(x0t, x1r, mrow, lrow);
  k_attn<<<dim3(T_ / 16, B_), 256, 0, stream>>>(x0t, x1r, x2b, mrow, lrow,
                                                Mbuf);
  k_out<<<dim3(T_ / 64, D_ / 64, B_), 128, 0, stream>>>(Mbuf, W4, b4, x, out);
}
